// ContTimeLSTM_26121991094631
// MI455X (gfx1250) — matrix operand order repaired and checked by reading, not compiled, not run on hardware
//
#include <hip/hip_runtime.h>

#define B_ 64
#define L_ 2048
#define DIN_ 256
#define H_ 256
#define KDIM 512          // DIN + H
#define NGROUPS 4         // batch groups of 16
#define NSPLIT 8          // WGs per group (column split)
#define COLS 32           // h-columns owned per WG
#define MW 16             // batches per group
#define TILES 14          // 7 gate blocks * (COLS/16)
#define KSTEPS 16         // KDIM / 32
#define WGSZ 512

typedef __bf16 bf16;
typedef bf16         v16bf __attribute__((ext_vector_type(16)));
typedef bf16         v8bf  __attribute__((ext_vector_type(8)));
typedef float        v8f   __attribute__((ext_vector_type(8)));
typedef float        v4f   __attribute__((ext_vector_type(4)));
typedef unsigned int v4u   __attribute__((ext_vector_type(4)));

// workspace byte offsets
#define WS_CNT   0        // 4 counters, 64B apart
#define WS_PROJ0 1024     // 1792 f32
#define WS_INIT  8192     // 4*256 f32: o0, cs0, ce0, d0
#define WS_HEX   12288    // [group][parity][m16][256] f32 h exchange (128 KB)

// dynamic LDS byte offsets
#define LDS_FRAG  0        // 14*16*32*16 bf16 = 229376 B (weights, WMMA-B layout)
#define LDS_INBUF 229376   // 16*512 bf16 = 16384 B  ([x_t | h] per batch)
#define LDS_PROJ  245760   // 7*32*16 f32 = 14336 B
#define LDS_BIAS  260096   // 224 f32
#define LDS_BYTES 262144

__device__ __forceinline__ unsigned short bfbits(float f) {
  unsigned int u = __builtin_bit_cast(unsigned int, f);
  unsigned int r = u + 0x7fffu + ((u >> 16) & 1u);
  return (unsigned short)(r >> 16);
}
__device__ __forceinline__ bf16 f2bf(float f) {
  unsigned short h = bfbits(f);
  return __builtin_bit_cast(bf16, h);
}
__device__ __forceinline__ unsigned int pack2(float a, float b) {
  return (unsigned int)bfbits(a) | ((unsigned int)bfbits(b) << 16);
}
__device__ __forceinline__ float sigm(float x) { return 1.f / (1.f + __expf(-x)); }
__device__ __forceinline__ float tanh_(float x) {
  float e = __expf(2.f * x);
  return 1.f - 2.f / (e + 1.f);
}
__device__ __forceinline__ float softplus_(float x) {
  float m = fmaxf(x, 0.f);
  return m + __logf(__expf(x - m) + __expf(-m));
}

// ---- prep: proj0 = start @ Wx.T + bias (K = 256) ----
__global__ void prep_proj0(const float* __restrict__ start,
                           const float* __restrict__ W,
                           const float* __restrict__ bias,
                           float* __restrict__ proj0) {
  int c = blockIdx.x * 256 + threadIdx.x;   // 7 blocks * 256 = 1792
  float acc = bias[c];
  const float* wr = W + (size_t)c * KDIM;
  #pragma unroll 8
  for (int k = 0; k < DIN_; ++k) acc += start[k] * wr[k];
  proj0[c] = acc;
}

// ---- prep: derive init state, zero group counters ----
__global__ void prep_init(unsigned char* ws) {
  int j = threadIdx.x;   // 0..255
  const float* p = (const float*)(ws + WS_PROJ0);
  float* init = (float*)(ws + WS_INIT);
  float ig  = sigm(p[j]);
  float ieg = sigm(p[2 * H_ + j]);
  float o0  = sigm(p[4 * H_ + j]);
  float z   = tanh_(p[5 * H_ + j]);
  init[j]          = o0;
  init[H_ + j]     = ig * z;                    // cs0
  init[2 * H_ + j] = ieg * z;                   // ce0
  init[3 * H_ + j] = softplus_(p[6 * H_ + j]);  // d0
  if (j < NGROUPS) *((unsigned int*)ws + j * 16) = 0u;
}

// ---- persistent fused recurrence ----
__global__ void __launch_bounds__(WGSZ, 1)
ctlstm_persistent(const float* __restrict__ x,
                  const float* __restrict__ td,
                  const float* __restrict__ W,
                  const float* __restrict__ bias,
                  float* __restrict__ outp,
                  unsigned char* __restrict__ ws) {
  extern __shared__ char smem[];
  bf16*  frag  = (bf16*)(smem + LDS_FRAG);
  bf16*  inb   = (bf16*)(smem + LDS_INBUF);
  float* projb = (float*)(smem + LDS_PROJ);
  float* biasb = (float*)(smem + LDS_BIAS);

  const int tid   = threadIdx.x;
  const int g     = blockIdx.x / NSPLIT;   // batch group
  const int wgig  = blockIdx.x % NSPLIT;   // column split index
  const int hbase = wgig * COLS;

  // One-time: weights -> LDS as bf16 WMMA-B fragments.
  // B(32x16) per (tile, kstep): lane l holds column (l&15), K = kstep*32 + (l>>4)*16 + i
  for (int e = tid; e < TILES * KSTEPS * 32 * 16; e += WGSZ) {
    int i    = e & 15;
    int lane = (e >> 4) & 31;
    int kk   = (e >> 9) & 15;
    int t    = e >> 13;
    int c    = (t >> 1) * H_ + hbase + ((t & 1) << 4) + (lane & 15);
    int k    = (kk << 5) + ((lane >> 4) << 4) + i;
    frag[e] = f2bf(W[(size_t)c * KDIM + k]);
  }
  for (int e = tid; e < 7 * COLS; e += WGSZ)
    biasb[e] = bias[(e >> 5) * H_ + hbase + (e & 31)];

  // carry state: one element per thread (m = batch-in-group, j = owned column)
  const int m = tid >> 5;
  const int j = tid & 31;
  const float* initp = (const float*)(ws + WS_INIT);
  float o  = initp[hbase + j];
  float cs = initp[H_ + hbase + j];
  float ce = initp[2 * H_ + hbase + j];
  float d  = initp[3 * H_ + hbase + j];

  unsigned int* cnt = (unsigned int*)ws + g * 16;
  const int b  = g * MW + m;
  const int pb = j * MW + m;   // projb per-thread index

  // x staging geometry: each thread owns 8 contiguous x elements of one batch row
  const int xm  = tid >> 5;          // batch-in-group
  const int xk0 = (tid & 31) << 3;   // k offset 0..248

  __syncthreads();

  #pragma unroll 1
  for (int s = 0; s < L_; ++s) {
    // 1) time decay + hidden output; publish own h element (f32, agent scope)
    float dt = td[(size_t)b * L_ + s];
    float c  = cs + (ce - cs) * __expf(-d * dt);
    float h  = o * tanh_(c);
    __builtin_nontemporal_store(h, outp + ((size_t)b * L_ + s) * H_ + hbase + j);
    float* hexf = (float*)(ws + WS_HEX) + (size_t)((g << 1) + (s & 1)) * (MW * H_);
    __hip_atomic_store(hexf + m * H_ + hbase + j, h,
                       __ATOMIC_RELEASE, __HIP_MEMORY_SCOPE_AGENT);

    // 2) stage x_t into LDS (vector loads, packed bf16 b128 store)
    {
      const v4f* xsrc = (const v4f*)(x + ((size_t)(g * MW + xm) * L_ + s) * DIN_ + xk0);
      v4f xa = __builtin_nontemporal_load(xsrc);
      v4f xb = __builtin_nontemporal_load(xsrc + 1);
      v4u val;
      val.x = pack2(xa.x, xa.y);
      val.y = pack2(xa.z, xa.w);
      val.z = pack2(xb.x, xb.y);
      val.w = pack2(xb.z, xb.w);
      *(v4u*)(inb + xm * KDIM + xk0) = val;
    }
    __threadfence();
    __syncthreads();

    // 3) group barrier: monotonic counter in L2 (double-buffered h by parity)
    if (tid == 0) {
      __hip_atomic_fetch_add(cnt, 1u, __ATOMIC_RELEASE, __HIP_MEMORY_SCOPE_AGENT);
      unsigned int tgt = (unsigned int)(NSPLIT * (s + 1));
      while (__hip_atomic_load(cnt, __ATOMIC_ACQUIRE, __HIP_MEMORY_SCOPE_AGENT) < tgt)
        __builtin_amdgcn_s_sleep(2);
    }
    __syncthreads();

    // 4) gather full h (256 cols x 16 batches), convert to bf16 into LDS
    {
      const float* hexf = (const float*)(ws + WS_HEX)
                        + (size_t)((g << 1) + (s & 1)) * (MW * H_);
      #pragma unroll
      for (int i = 0; i < 4; ++i) {
        int p  = tid + (i << 9);          // word-pair index 0..2047
        int mm = p >> 7, k0 = (p & 127) << 1;
        float f0 = __hip_atomic_load(hexf + mm * H_ + k0,
                                     __ATOMIC_RELAXED, __HIP_MEMORY_SCOPE_AGENT);
        float f1 = __hip_atomic_load(hexf + mm * H_ + k0 + 1,
                                     __ATOMIC_RELAXED, __HIP_MEMORY_SCOPE_AGENT);
        *(unsigned int*)(inb + mm * KDIM + DIN_ + k0) = pack2(f0, f1);
      }
    }
    __syncthreads();

    // 5) GEMM: one 16x16 proj tile per wave, K=512 via 16x v_wmma_f32_16x16x32_bf16
    int wv = tid >> 5;
    if (wv < TILES) {
      int lane  = tid & 31;
      int block = wv >> 1;
      int jb    = (wv & 1) << 4;
      int n     = lane & 15;
      int koff  = (lane >> 4) << 4;                  // A K-offset per lane half
      const bf16* arow = inb + (lane & 15) * KDIM;   // A row = batch (lane&15)
      v8f acc = {};
      #pragma unroll
      for (int kk = 0; kk < KSTEPS; ++kk) {
        v8bf a0 = *(const v8bf*)(arow + (kk << 5) + koff);
        v8bf a1 = *(const v8bf*)(arow + (kk << 5) + 8 + koff);
        v16bf av;
        #pragma unroll
        for (int q2 = 0; q2 < 8; ++q2) { av[q2] = a0[q2]; av[q2 + 8] = a1[q2]; }
        v16bf bv = *(const v16bf*)(frag + wv * 8192 + (kk << 9) + (lane << 4));
        acc = __builtin_amdgcn_wmma_f32_16x16x32_bf16(false, av, false, bv,
                                                      (short)0, acc, false, false);
      }
      float bb = biasb[block * COLS + jb + n];
      int mr = (lane >> 4) << 3;   // C: VGPR r holds row r (+8 for upper lanes)
      #pragma unroll
      for (int r = 0; r < 8; ++r)
        projb[(block * COLS + jb + n) * MW + mr + r] = acc[r] + bb;
    }
    __syncthreads();

    // 6) gates + state update + emit state block (streaming stores)
    float ig  = sigm(projb[pb]);
    float fg  = sigm(projb[pb + 1 * COLS * MW]);
    float ieg = sigm(projb[pb + 2 * COLS * MW]);
    float feg = sigm(projb[pb + 3 * COLS * MW]);
    float on  = sigm(projb[pb + 4 * COLS * MW]);
    float z   = tanh_(projb[pb + 5 * COLS * MW]);
    float dn  = softplus_(projb[pb + 6 * COLS * MW]);
    float csn = fg * c + ig * z;
    float cen = feg * ce + ieg * z;
    size_t sb = (size_t)B_ * L_ * H_ + ((size_t)b * L_ + s) * (4 * H_) + hbase + j;
    __builtin_nontemporal_store(on,  outp + sb);
    __builtin_nontemporal_store(csn, outp + sb + H_);
    __builtin_nontemporal_store(cen, outp + sb + 2 * H_);
    __builtin_nontemporal_store(dn,  outp + sb + 3 * H_);
    o = on; cs = csn; ce = cen; d = dn;
  }
}

extern "C" void kernel_launch(void* const* d_in, const int* in_sizes, int n_in,
                              void* d_out, int out_size, void* d_ws, size_t ws_size,
                              hipStream_t stream) {
  (void)in_sizes; (void)n_in; (void)out_size; (void)ws_size;
  const float* x     = (const float*)d_in[0];
  const float* tdel  = (const float*)d_in[1];
  const float* start = (const float*)d_in[2];
  const float* W     = (const float*)d_in[3];
  const float* bias  = (const float*)d_in[4];
  float* outp = (float*)d_out;
  unsigned char* ws = (unsigned char*)d_ws;

  prep_proj0<<<7, 256, 0, stream>>>(start, W, bias, (float*)(ws + WS_PROJ0));
  prep_init<<<1, 256, 0, stream>>>(ws);
  ctlstm_persistent<<<NGROUPS * NSPLIT, WGSZ, LDS_BYTES, stream>>>(
      x, tdel, W, bias, outp, ws);
}
